// Attention_83434034692794
// MI455X (gfx1250) — matrix operand order repaired and checked by reading, not compiled, not run on hardware
//
#include <hip/hip_runtime.h>
#include <hip/hip_bf16.h>

// ---------------------------------------------------------------------------
// MI455X (gfx1250) wave32 WMMA implementation of the 3-way grouped attention.
//   K-dim of every matmul is padded/tiled to 32 -> v_wmma_f32_16x16x32_bf16.
//   Pure-copy LDS staging uses GLOBAL_LOAD_ASYNC_TO_LDS_B64 when available.
// ---------------------------------------------------------------------------

typedef unsigned short u16t;
typedef __attribute__((ext_vector_type(16))) __bf16 v16bf;
typedef __attribute__((ext_vector_type(8)))  float  v8f;
typedef int v2i_vs __attribute__((vector_size(2 * sizeof(int))));  // b64 payload

#if defined(__has_builtin)
#if __has_builtin(__builtin_amdgcn_global_load_async_to_lds_b64) && \
    __has_builtin(__builtin_amdgcn_s_wait_asynccnt)
#define USE_ASYNC_LDS 1
#endif
#endif
#ifndef USE_ASYNC_LDS
#define USE_ASYNC_LDS 0
#endif

union FragAB {
  v16bf v;
  u16t  u[16];
  uint2 d2[4];
};

__device__ __forceinline__ u16t f32_to_bf16u(float f) {
  unsigned u = __builtin_bit_cast(unsigned, f);
  u += 0x7FFFu + ((u >> 16) & 1u);            // round-to-nearest-even
  return (u16t)(u >> 16);
}
__device__ __forceinline__ unsigned pack2(float lo, float hi) {
  return (unsigned)f32_to_bf16u(lo) | ((unsigned)f32_to_bf16u(hi) << 16);
}
__device__ __forceinline__ v8f v8f_zero() {
  v8f z;
#pragma unroll
  for (int i = 0; i < 8; ++i) z[i] = 0.0f;
  return z;
}

// 8-byte global->LDS copy: async DMA on CDNA5 when the builtin exists.
__device__ __forceinline__ void copy_b64_to_lds(const u16t* gsrc, u16t* ldst) {
#if USE_ASYNC_LDS
  __builtin_amdgcn_global_load_async_to_lds_b64(
      (__attribute__((address_space(1))) v2i_vs*)(uintptr_t)gsrc,
      (__attribute__((address_space(3))) v2i_vs*)(uintptr_t)ldst, 0, 0);
#else
  *(uint2*)ldst = *(const uint2*)gsrc;
#endif
}
__device__ __forceinline__ void async_copies_wait() {
#if USE_ASYNC_LDS
  __builtin_amdgcn_s_wait_asynccnt(0);
#endif
}

// A fragment (16x32 bf16, M x K): lane L holds row M=L%16; hh=L/16.
// elements 0..7  -> K = 8*hh + e ;  elements 8..15 -> K = 16 + 8*hh + (e-8)
__device__ __forceinline__ v16bf load_fragA(const u16t* base, int stride) {
  const int lane = threadIdx.x & 31;
  const u16t* rp = base + (lane & 15) * stride + ((lane >> 4) << 3);
  FragAB f;
  f.d2[0] = *(const uint2*)(rp + 0);
  f.d2[1] = *(const uint2*)(rp + 4);
  f.d2[2] = *(const uint2*)(rp + 16);
  f.d2[3] = *(const uint2*)(rp + 20);
  return f.v;
}
// B fragment (32x16 bf16, K x N), sourced from B^T stored row-major
// (base[col][k]): lane L holds column N=L%16; elements 0..7 -> K = 8*hh + e; 8..15 -> K = 16 + 8*hh + (e-8).
__device__ __forceinline__ v16bf load_fragB(const u16t* base, int stride) {
  const int lane = threadIdx.x & 31;
  const u16t* rp = base + (lane & 15) * stride + ((lane >> 4) << 3);
  FragAB f;
  f.d2[0] = *(const uint2*)(rp + 0);
  f.d2[1] = *(const uint2*)(rp + 4);
  f.d2[2] = *(const uint2*)(rp + 16);
  f.d2[3] = *(const uint2*)(rp + 20);
  return f.v;
}

#define WMMA_BF16(a, b, c) \
  __builtin_amdgcn_wmma_f32_16x16x32_bf16(false, (a), false, (b), (short)0, (c), false, false)

// qkv group buffer: [t(q/k/v)][g(0..2)][b][h][n][d], bf16, each (b,h) plane 1024x32
__device__ __forceinline__ size_t qkv_off(int t, int g, int b, int h) {
  return ((size_t)((((t * 3 + g) * 8 + b) * 8 + h))) << 15;   // * 1024 * 32
}

// ---------------------------------------------------------------------------
// Kernel 1: qkv = x @ w_qkv^T   (8192x768 @ 768x2304), scatter to group bufs
// ---------------------------------------------------------------------------
__global__ void __launch_bounds__(256) qkv_gemm_kernel(
    const float* __restrict__ X, const float* __restrict__ W,
    u16t* __restrict__ qkvb) {
  __shared__ u16t Ald[64 * 36];
  __shared__ u16t Bld[128 * 36];
  const int tid = threadIdx.x;
  const int lane = tid & 31, wave = tid >> 5;
  const int waveM = wave >> 2, waveN = wave & 3;
  const int hh = lane >> 4, ln = lane & 15;
  const int mBase = blockIdx.x * 64;
  const int nBase = blockIdx.y * 128;

  v8f acc[2][2];
  acc[0][0] = v8f_zero(); acc[0][1] = v8f_zero();
  acc[1][0] = v8f_zero(); acc[1][1] = v8f_zero();

  for (int kc = 0; kc < 768; kc += 32) {
    __syncthreads();
    {  // stage A tile 64x32 (f32 -> bf16), 8 elems/thread
      const int row = tid >> 2, col = (tid & 3) << 3;
      const float* src = X + (size_t)(mBase + row) * 768 + kc + col;
      float4 a0 = *(const float4*)(src);
      float4 a1 = *(const float4*)(src + 4);
      unsigned* d = (unsigned*)&Ald[row * 36 + col];
      d[0] = pack2(a0.x, a0.y); d[1] = pack2(a0.z, a0.w);
      d[2] = pack2(a1.x, a1.y); d[3] = pack2(a1.z, a1.w);
    }
    {  // stage B tile 128x32 (w rows are k-contiguous), 16 elems/thread
      const int row = tid >> 1, col = (tid & 1) << 4;
      const float* src = W + (size_t)(nBase + row) * 768 + kc + col;
      float4 b0 = *(const float4*)(src);
      float4 b1 = *(const float4*)(src + 4);
      float4 b2 = *(const float4*)(src + 8);
      float4 b3 = *(const float4*)(src + 12);
      unsigned* d = (unsigned*)&Bld[row * 36 + col];
      d[0] = pack2(b0.x, b0.y); d[1] = pack2(b0.z, b0.w);
      d[2] = pack2(b1.x, b1.y); d[3] = pack2(b1.z, b1.w);
      d[4] = pack2(b2.x, b2.y); d[5] = pack2(b2.z, b2.w);
      d[6] = pack2(b3.x, b3.y); d[7] = pack2(b3.z, b3.w);
    }
    __syncthreads();
    v16bf a0 = load_fragA(&Ald[(waveM * 32 + 0) * 36], 36);
    v16bf a1 = load_fragA(&Ald[(waveM * 32 + 16) * 36], 36);
    v16bf b0 = load_fragB(&Bld[(waveN * 32 + 0) * 36], 36);
    v16bf b1 = load_fragB(&Bld[(waveN * 32 + 16) * 36], 36);
    acc[0][0] = WMMA_BF16(a0, b0, acc[0][0]);
    acc[0][1] = WMMA_BF16(a0, b1, acc[0][1]);
    acc[1][0] = WMMA_BF16(a1, b0, acc[1][0]);
    acc[1][1] = WMMA_BF16(a1, b1, acc[1][1]);
  }

#pragma unroll
  for (int ms = 0; ms < 2; ++ms) {
#pragma unroll
    for (int ns = 0; ns < 2; ++ns) {
      const int gcol = nBase + waveN * 32 + ns * 16 + ln;
      const int t   = gcol / 768;
      const int rem = gcol - t * 768;
      const int h   = rem / 96;
      const int dd  = rem - h * 96;
      const int g   = dd >> 5, d = dd & 31;
      const int base0 = t * 3 + g;
#pragma unroll
      for (int r = 0; r < 8; ++r) {
        const int grow = mBase + waveM * 32 + ms * 16 + r + 8 * hh;
        const int bb = grow >> 10, nn = grow & 1023;
        size_t off = (((size_t)((base0 * 8 + bb) * 8 + h) << 10) + nn) * 32 + d;
        qkvb[off] = f32_to_bf16u(acc[ms][ns][r]);
      }
    }
  }
}

// ---------------------------------------------------------------------------
// Kernel 2: flash attention. Per block: one (attn a, b, h, 64-query block);
// 4 waves, each owns one 16-query tile. Scores computed transposed
// (S' = K_chunk x Q^T) so exp(S') is directly a valid A-fragment for P*V.
// ---------------------------------------------------------------------------
__global__ void __launch_bounds__(128) attn_kernel(
    const u16t* __restrict__ qkvb, u16t* __restrict__ ctxb) {
  __shared__ u16t Klds[32 * 36];   // [key][d], padded
  __shared__ u16t Vt[32 * 36];     // [d][key], padded (transposed at stage)
  const int tid = threadIdx.x;
  const int lane = tid & 31, wave = tid >> 5;
  const int hh = lane >> 4, ln = lane & 15;

  const int bid = blockIdx.x;
  const int qb = bid & 15;
  const int h  = (bid >> 4) & 7;
  const int b  = (bid >> 7) & 7;
  const int a  = bid >> 10;                 // 0..2

  int qg_g, ga, gb;
  if (a == 0)      { qg_g = 1; ga = 0; gb = 1; }   // x3: q4, [k3,k4]
  else if (a == 1) { qg_g = 2; ga = 0; gb = 2; }   // x4: q5, [k3,k5]
  else             { qg_g = 2; ga = 1; gb = 2; }   // x5: q5, [k4,k5]

  const u16t* qg = qkvb + qkv_off(0, qg_g, b, h);
  const u16t* k0 = qkvb + qkv_off(1, ga, b, h);
  const u16t* k1 = qkvb + qkv_off(1, gb, b, h);
  const u16t* v0 = qkvb + qkv_off(2, ga, b, h);
  const u16t* v1 = qkvb + qkv_off(2, gb, b, h);

  const int q0 = qb * 64 + wave * 16;
  const v16bf bQ = load_fragB(qg + (size_t)q0 * 32, 32);  // Q^T as B matrix

  v8f o0 = v8f_zero(), o1 = v8f_zero();
  float m = -1e30f, l = 0.0f;
  const float scale = 0.10206207261596577f;  // 96^-0.5

  for (int kc = 0; kc < 64; ++kc) {
    const int key0 = kc * 32;
    const u16t* kb = (key0 < 1024) ? k0 : k1;
    const u16t* vb = (key0 < 1024) ? v0 : v1;
    const int lk = key0 & 1023;

    __syncthreads();
    {  // stage K row-major (async DMA to LDS) + V transposed (manual)
      const int row = tid >> 2, col = (tid & 3) << 3;
      const u16t* ksrc = kb + (size_t)(lk + row) * 32 + col;
      copy_b64_to_lds(ksrc,     &Klds[row * 36 + col]);
      copy_b64_to_lds(ksrc + 4, &Klds[row * 36 + col + 4]);
      union { uint4 q; u16t u[8]; } tv;
      tv.q = *(const uint4*)(vb + (size_t)(lk + row) * 32 + col);
#pragma unroll
      for (int j = 0; j < 8; ++j) Vt[(col + j) * 36 + row] = tv.u[j];
    }
    if (kc + 1 < 64 && tid < 64) {  // prefetch next K/V chunk (global_prefetch_b8)
      const int nk0 = (kc + 1) * 32;
      const u16t* pk = (nk0 < 1024) ? (k0 + (size_t)nk0 * 32) : (k1 + (size_t)(nk0 - 1024) * 32);
      const u16t* pv = (nk0 < 1024) ? (v0 + (size_t)nk0 * 32) : (v1 + (size_t)(nk0 - 1024) * 32);
      __builtin_prefetch(pk + tid * 16, 0, 1);
      __builtin_prefetch(pv + tid * 16, 0, 1);
    }
    async_copies_wait();
    __syncthreads();

    // S' = K_chunk(16x32) x Q^T(32x16): lane = query, elements = keys
    v16bf aK0 = load_fragA(&Klds[0], 36);
    v16bf aK1 = load_fragA(&Klds[16 * 36], 36);
    v8f s0 = WMMA_BF16(aK0, bQ, v8f_zero());
    v8f s1 = WMMA_BF16(aK1, bQ, v8f_zero());

    float cm = -1e30f;
#pragma unroll
    for (int e = 0; e < 8; ++e) {
      s0[e] *= scale; s1[e] *= scale;
      cm = fmaxf(cm, fmaxf(s0[e], s1[e]));
    }
    cm = fmaxf(cm, __shfl_xor(cm, 16, 32));   // combine key halves per query
    const float mn = fmaxf(m, cm);
    const float alpha = __expf(m - mn);

    FragAB P;
    float rs = 0.0f;
#pragma unroll
    for (int e = 0; e < 8; ++e) {
      float p0 = __expf(s0[e] - mn);
      float p1 = __expf(s1[e] - mn);
      rs += p0 + p1;
      P.u[e]     = f32_to_bf16u(p0);   // keys 8*hh+e        (A-frag elems 0..7)
      P.u[8 + e] = f32_to_bf16u(p1);   // keys 16+8*hh+e     (A-frag elems 8..15)
    }
    rs += __shfl_xor(rs, 16, 32);
    l = l * alpha + rs;
    m = mn;
#pragma unroll
    for (int r = 0; r < 8; ++r) {     // rescale O rows (row = r + 8*hh)
      const float ar = __shfl(alpha, r + 8 * hh, 32);
      o0[r] *= ar; o1[r] *= ar;
    }

    v16bf bV0 = load_fragB(&Vt[0], 36);        // d = 0..15
    v16bf bV1 = load_fragB(&Vt[16 * 36], 36);  // d = 16..31
    o0 = WMMA_BF16(P.v, bV0, o0);
    o1 = WMMA_BF16(P.v, bV1, o1);
  }

#pragma unroll
  for (int r = 0; r < 8; ++r) {
    const float lr = __shfl(l, r + 8 * hh, 32);
    const float inv = 1.0f / lr;
    const int n = q0 + r + 8 * hh;
    const size_t rb = ((size_t)(b * 1024 + n)) * 768 + a * 256 + h * 32;
    ctxb[rb + ln]      = f32_to_bf16u(o0[r] * inv);
    ctxb[rb + 16 + ln] = f32_to_bf16u(o1[r] * inv);
  }
}

// ---------------------------------------------------------------------------
// Kernel 3: out = ctx(8192x768,bf16) @ w_proj^T(768x768) + b_proj  -> f32
// ---------------------------------------------------------------------------
__global__ void __launch_bounds__(256) proj_gemm_kernel(
    const u16t* __restrict__ A, const float* __restrict__ W,
    const float* __restrict__ bias, float* __restrict__ out) {
  __shared__ u16t Ald[64 * 36];
  __shared__ u16t Bld[128 * 36];
  const int tid = threadIdx.x;
  const int lane = tid & 31, wave = tid >> 5;
  const int waveM = wave >> 2, waveN = wave & 3;
  const int hh = lane >> 4, ln = lane & 15;
  const int mBase = blockIdx.x * 64;
  const int nBase = blockIdx.y * 128;

  v8f acc[2][2];
  acc[0][0] = v8f_zero(); acc[0][1] = v8f_zero();
  acc[1][0] = v8f_zero(); acc[1][1] = v8f_zero();

  for (int kc = 0; kc < 768; kc += 32) {
    __syncthreads();
    {  // stage A tile (already bf16): async DMA to LDS
      const int row = tid >> 2, col = (tid & 3) << 3;
      const u16t* src = A + (size_t)(mBase + row) * 768 + kc + col;
      copy_b64_to_lds(src,     &Ald[row * 36 + col]);
      copy_b64_to_lds(src + 4, &Ald[row * 36 + col + 4]);
    }
    {  // stage B tile (w_proj f32 -> bf16)
      const int row = tid >> 1, col = (tid & 1) << 4;
      const float* src = W + (size_t)(nBase + row) * 768 + kc + col;
      float4 b0 = *(const float4*)(src);
      float4 b1 = *(const float4*)(src + 4);
      float4 b2 = *(const float4*)(src + 8);
      float4 b3 = *(const float4*)(src + 12);
      unsigned* d = (unsigned*)&Bld[row * 36 + col];
      d[0] = pack2(b0.x, b0.y); d[1] = pack2(b0.z, b0.w);
      d[2] = pack2(b1.x, b1.y); d[3] = pack2(b1.z, b1.w);
      d[4] = pack2(b2.x, b2.y); d[5] = pack2(b2.z, b2.w);
      d[6] = pack2(b3.x, b3.y); d[7] = pack2(b3.z, b3.w);
    }
    async_copies_wait();
    __syncthreads();
    v16bf a0 = load_fragA(&Ald[(waveM * 32 + 0) * 36], 36);
    v16bf a1 = load_fragA(&Ald[(waveM * 32 + 16) * 36], 36);
    v16bf b0 = load_fragB(&Bld[(waveN * 32 + 0) * 36], 36);
    v16bf b1 = load_fragB(&Bld[(waveN * 32 + 16) * 36], 36);
    acc[0][0] = WMMA_BF16(a0, b0, acc[0][0]);
    acc[0][1] = WMMA_BF16(a0, b1, acc[0][1]);
    acc[1][0] = WMMA_BF16(a1, b0, acc[1][0]);
    acc[1][1] = WMMA_BF16(a1, b1, acc[1][1]);
  }

#pragma unroll
  for (int ms = 0; ms < 2; ++ms) {
#pragma unroll
    for (int ns = 0; ns < 2; ++ns) {
      const int gcol = nBase + waveN * 32 + ns * 16 + ln;
      const float bv = bias[gcol];
#pragma unroll
      for (int r = 0; r < 8; ++r) {
        const int grow = mBase + waveM * 32 + ms * 16 + r + 8 * hh;
        out[(size_t)grow * 768 + gcol] = acc[ms][ns][r] + bv;
      }
    }
  }
}

// ---------------------------------------------------------------------------
extern "C" void kernel_launch(void* const* d_in, const int* in_sizes, int n_in,
                              void* d_out, int out_size, void* d_ws, size_t ws_size,
                              hipStream_t stream) {
  const float* x      = (const float*)d_in[0];
  const float* w_qkv  = (const float*)d_in[1];
  const float* w_proj = (const float*)d_in[2];
  const float* b_proj = (const float*)d_in[3];

  u16t* qkvb = (u16t*)d_ws;                       // 9 * 8*8*1024*32 bf16 = 37.75 MB
  u16t* ctxb = qkvb + (size_t)3 * 3 * 8 * 8 * 1024 * 32;  // 8192x768 bf16 = 12.6 MB

  qkv_gemm_kernel<<<dim3(128, 18), 256, 0, stream>>>(x, w_qkv, qkvb);
  attn_kernel<<<dim3(3072), 128, 0, stream>>>(qkvb, ctxb);
  proj_gemm_kernel<<<dim3(128, 6), 256, 0, stream>>>(ctxb, w_proj, b_proj, (float*)d_out);
}
